// ClusteringDistribution_20091857010845
// MI455X (gfx1250) — matrix operand order repaired and checked by reading, not compiled, not run on hardware
//
#include <hip/hip_runtime.h>
#include <hip/hip_bf16.h>
#include <stdint.h>

// Problem sizes (fixed by the reference)
#define NROWS 32768   // features rows
#define KCLUS 2048    // prototypes / output columns
#define DDIM  512     // feature dimension

typedef __attribute__((ext_vector_type(16))) __bf16 v16bf;
typedef __attribute__((ext_vector_type(8)))  __bf16 v8bf;
typedef __attribute__((ext_vector_type(8)))  float  v8f;

// ---------- bf16 helpers (round-to-nearest-even) ----------
__device__ __forceinline__ void split_f32(float f, uint16_t& h, uint16_t& l) {
    uint32_t u  = __builtin_bit_cast(uint32_t, f);
    uint32_t hr = (u + 0x7FFFu + ((u >> 16) & 1u)) >> 16;
    float    hf = __builtin_bit_cast(float, hr << 16);
    float    r  = f - hf;
    uint32_t ru = __builtin_bit_cast(uint32_t, r);
    uint32_t lr = (ru + 0x7FFFu + ((ru >> 16) & 1u)) >> 16;
    h = (uint16_t)hr;
    l = (uint16_t)lr;
}
__device__ __forceinline__ __bf16 f32_to_bf16(float f) {
    uint32_t u = __builtin_bit_cast(uint32_t, f);
    uint32_t r = (u + 0x7FFFu + ((u >> 16) & 1u)) >> 16;
    return __builtin_bit_cast(__bf16, (uint16_t)r);
}
__device__ __forceinline__ float bf16_to_f32(__bf16 b) {
    uint32_t u = ((uint32_t)__builtin_bit_cast(uint16_t, b)) << 16;
    return __builtin_bit_cast(float, u);
}

// ---------- kernel 1: f32 -> (hi, lo) bf16 planes + squared row norms ----------
// One wave32 per row; 8 rows per 256-thread block. Rows [0, NROWS) = features,
// rows [NROWS, NROWS+KCLUS) = prototypes.
__global__ __launch_bounds__(256) void convert_norms_kernel(
    const float* __restrict__ F, const float* __restrict__ P,
    float* __restrict__ f2, float* __restrict__ p2,
    uint16_t* __restrict__ Fhi, uint16_t* __restrict__ Flo,
    uint16_t* __restrict__ Phi, uint16_t* __restrict__ Plo) {
    const int lane = threadIdx.x & 31;
    const int wave = threadIdx.x >> 5;
    const int row  = blockIdx.x * 8 + wave;

    const float* src;
    float* dst;
    uint16_t *hp, *lp;
    if (row < NROWS) {
        const size_t off = (size_t)row * DDIM;
        src = F + off; dst = f2 + row; hp = Fhi + off; lp = Flo + off;
    } else {
        const int pr = row - NROWS;
        const size_t off = (size_t)pr * DDIM;
        src = P + off; dst = p2 + pr; hp = Phi + off; lp = Plo + off;
    }

    float s = 0.0f;
#pragma unroll
    for (int i = 0; i < 4; ++i) {
        const int e = 4 * (lane + 32 * i);
        float4 x = *(const float4*)(src + e);
        uint16_t h0, h1, h2, h3, l0, l1, l2, l3;
        split_f32(x.x, h0, l0);
        split_f32(x.y, h1, l1);
        split_f32(x.z, h2, l2);
        split_f32(x.w, h3, l3);
        uint2 hv, lv;
        hv.x = (uint32_t)h0 | ((uint32_t)h1 << 16);
        hv.y = (uint32_t)h2 | ((uint32_t)h3 << 16);
        lv.x = (uint32_t)l0 | ((uint32_t)l1 << 16);
        lv.y = (uint32_t)l2 | ((uint32_t)l3 << 16);
        *(uint2*)(hp + e) = hv;
        *(uint2*)(lp + e) = lv;
        s += x.x * x.x + x.y * x.y + x.z * x.z + x.w * x.w;
    }
#pragma unroll
    for (int off = 16; off > 0; off >>= 1) s += __shfl_down(s, off, 32);
    if (lane == 0) *dst = s;
}

// Norms-only kernel (fallback path)
__global__ __launch_bounds__(256) void norms_kernel(
    const float* __restrict__ F, const float* __restrict__ P,
    float* __restrict__ f2, float* __restrict__ p2) {
    const int lane = threadIdx.x & 31;
    const int wave = threadIdx.x >> 5;
    const int row  = blockIdx.x * 8 + wave;
    const float* src;
    float* dst;
    if (row < NROWS) { src = F + (size_t)row * DDIM; dst = f2 + row; }
    else { const int pr = row - NROWS; src = P + (size_t)pr * DDIM; dst = p2 + pr; }
    float s = 0.0f;
#pragma unroll
    for (int i = 0; i < 4; ++i) {
        float4 x = *(const float4*)(src + 4 * (lane + 32 * i));
        s += x.x * x.x + x.y * x.y + x.z * x.z + x.w * x.w;
    }
#pragma unroll
    for (int off = 16; off > 0; off >>= 1) s += __shfl_down(s, off, 32);
    if (lane == 0) *dst = s;
}

// A-fragment loader: elems 0-7 at p, elems 8-15 at p+16 (both 16B loads).
__device__ __forceinline__ v16bf load_afrag(const uint16_t* p) {
    v8bf a = *(const v8bf*)(p);
    v8bf b = *(const v8bf*)(p + 16);
    return __builtin_shufflevector(a, b, 0, 1, 2, 3, 4, 5, 6, 7,
                                         8, 9, 10, 11, 12, 13, 14, 15);
}

// ---------- kernel 2: WMMA GEMM on prebuilt bf16 planes, fused sim epilogue ----------
// Block: 256 threads = 8 waves, arranged 2(M) x 4(N); each wave owns 32x64
// (2x4 WMMA tiles). Block covers 64(M) x 256(N).
__global__ __launch_bounds__(256) void gemm_sim_planes_kernel(
    const uint16_t* __restrict__ Fhi, const uint16_t* __restrict__ Flo,
    const uint16_t* __restrict__ Phi, const uint16_t* __restrict__ Plo,
    const float* __restrict__ f2, const float* __restrict__ p2,
    float* __restrict__ out) {
    const int lane = threadIdx.x & 31;
    const int wave = threadIdx.x >> 5;
    const int half = lane >> 4;
    const int l15  = lane & 15;

    const int mBw = blockIdx.y * 64  + (wave & 1) * 32;
    const int nBw = blockIdx.x * 256 + (wave >> 1) * 64;

    v8f zero;
#pragma unroll
    for (int i = 0; i < 8; ++i) zero[i] = 0.0f;
    v8f c[2][4];
#pragma unroll
    for (int mi = 0; mi < 2; ++mi)
#pragma unroll
        for (int ni = 0; ni < 4; ++ni) c[mi][ni] = zero;

    for (int k0 = 0; k0 < DDIM; k0 += 32) {
        // A fragments: lanes 0-15 hold row M=l15, K 0-7 (elems 0-7) and
        // K 16-23 (elems 8-15); lanes 16-31 hold K 8-15 / 24-31.
        v16bf ah[2], al[2];
#pragma unroll
        for (int mi = 0; mi < 2; ++mi) {
            const size_t ao = (size_t)(mBw + mi * 16 + l15) * DDIM + k0 + half * 8;
            ah[mi] = load_afrag(Fhi + ao);
            al[mi] = load_afrag(Flo + ao);
        }
        // B fragments (B = prototypes^T): lanes 0-15 hold column N=l15,
        // K 0-7 / 16-23; lanes 16-31 hold K 8-15 / 24-31 (same as A).
        v16bf bh[4], bl[4];
#pragma unroll
        for (int ni = 0; ni < 4; ++ni) {
            const size_t bo = (size_t)(nBw + ni * 16 + l15) * DDIM + k0 + half * 8;
            bh[ni] = load_afrag(Phi + bo);
            bl[ni] = load_afrag(Plo + bo);
        }
        // fp ~= hi*hi + hi*lo + lo*hi
#pragma unroll
        for (int mi = 0; mi < 2; ++mi)
#pragma unroll
            for (int ni = 0; ni < 4; ++ni) {
                c[mi][ni] = __builtin_amdgcn_wmma_f32_16x16x32_bf16(
                    false, ah[mi], false, bh[ni], (short)0, c[mi][ni], false, false);
                c[mi][ni] = __builtin_amdgcn_wmma_f32_16x16x32_bf16(
                    false, ah[mi], false, bl[ni], (short)0, c[mi][ni], false, false);
                c[mi][ni] = __builtin_amdgcn_wmma_f32_16x16x32_bf16(
                    false, al[mi], false, bh[ni], (short)0, c[mi][ni], false, false);
            }
    }

    // Epilogue: C/D layout — VGPR j: lanes 0-15 => M=j, lanes 16-31 => M=j+8,
    // N = lane & 15.  sim = 1 / (1 + sqrt(max(f2 + p2 - 2 fp, 0)))
#pragma unroll
    for (int mi = 0; mi < 2; ++mi)
#pragma unroll
        for (int ni = 0; ni < 4; ++ni) {
            const int n = nBw + ni * 16 + l15;
            const float pn = p2[n];
#pragma unroll
            for (int j = 0; j < 8; ++j) {
                const int m = mBw + mi * 16 + half * 8 + j;
                float sq = f2[m] + pn - 2.0f * c[mi][ni][j];
                sq = sq > 0.0f ? sq : 0.0f;
                out[(size_t)m * KCLUS + n] = 1.0f / (1.0f + sqrtf(sq));
            }
        }
}

// ---------- fallback kernel: inline-split GEMM (only if ws too small) ----------
__device__ __forceinline__ void load_split8(const float* __restrict__ p,
                                            v16bf& hi, v16bf& lo, int base) {
    float4 x0 = *(const float4*)(p);
    float4 x1 = *(const float4*)(p + 4);
    float v[8] = {x0.x, x0.y, x0.z, x0.w, x1.x, x1.y, x1.z, x1.w};
#pragma unroll
    for (int i = 0; i < 8; ++i) {
        __bf16 h = f32_to_bf16(v[i]);
        hi[base + i] = h;
        lo[base + i] = f32_to_bf16(v[i] - bf16_to_f32(h));
    }
}

__global__ __launch_bounds__(256) void gemm_sim_inline_kernel(
    const float* __restrict__ F, const float* __restrict__ P,
    const float* __restrict__ f2, const float* __restrict__ p2,
    float* __restrict__ out) {
    const int lane = threadIdx.x & 31;
    const int wave = threadIdx.x >> 5;
    const int half = lane >> 4;
    const int l15  = lane & 15;
    const int mBw = blockIdx.y * 64  + (wave & 1) * 32;
    const int nBw = blockIdx.x * 128 + (wave >> 1) * 32;

    v8f zero;
#pragma unroll
    for (int i = 0; i < 8; ++i) zero[i] = 0.0f;
    v8f c[2][2];
#pragma unroll
    for (int mi = 0; mi < 2; ++mi)
#pragma unroll
        for (int ni = 0; ni < 2; ++ni) c[mi][ni] = zero;

    for (int k0 = 0; k0 < DDIM; k0 += 32) {
        v16bf ahi[2], alo[2], bhi[2], blo[2];
#pragma unroll
        for (int mi = 0; mi < 2; ++mi) {
            const float* ap = F + (size_t)(mBw + mi * 16 + l15) * DDIM + k0 + half * 8;
            load_split8(ap,      ahi[mi], alo[mi], 0);
            load_split8(ap + 16, ahi[mi], alo[mi], 8);
        }
#pragma unroll
        for (int ni = 0; ni < 2; ++ni) {
            const float* bp = P + (size_t)(nBw + ni * 16 + l15) * DDIM + k0 + half * 8;
            load_split8(bp,      bhi[ni], blo[ni], 0);
            load_split8(bp + 16, bhi[ni], blo[ni], 8);
        }
#pragma unroll
        for (int mi = 0; mi < 2; ++mi)
#pragma unroll
            for (int ni = 0; ni < 2; ++ni) {
                c[mi][ni] = __builtin_amdgcn_wmma_f32_16x16x32_bf16(
                    false, ahi[mi], false, bhi[ni], (short)0, c[mi][ni], false, false);
                c[mi][ni] = __builtin_amdgcn_wmma_f32_16x16x32_bf16(
                    false, ahi[mi], false, blo[ni], (short)0, c[mi][ni], false, false);
                c[mi][ni] = __builtin_amdgcn_wmma_f32_16x16x32_bf16(
                    false, alo[mi], false, bhi[ni], (short)0, c[mi][ni], false, false);
            }
    }
#pragma unroll
    for (int mi = 0; mi < 2; ++mi)
#pragma unroll
        for (int ni = 0; ni < 2; ++ni) {
            const int n = nBw + ni * 16 + l15;
            const float pn = p2[n];
#pragma unroll
            for (int j = 0; j < 8; ++j) {
                const int m = mBw + mi * 16 + half * 8 + j;
                float sq = f2[m] + pn - 2.0f * c[mi][ni][j];
                sq = sq > 0.0f ? sq : 0.0f;
                out[(size_t)m * KCLUS + n] = 1.0f / (1.0f + sqrtf(sq));
            }
        }
}

// ---------- kernel 3: in-place row softmax over 2048 columns ----------
__global__ __launch_bounds__(256) void softmax_kernel(float* __restrict__ out) {
    __shared__ float red[256];
    const int tid = threadIdx.x;
    float* rp = out + (size_t)blockIdx.x * KCLUS;

    float v[8];
    float mx = -3.0e38f;
#pragma unroll
    for (int i = 0; i < 8; ++i) {
        v[i] = rp[tid + i * 256];
        mx = fmaxf(mx, v[i]);
    }
    red[tid] = mx;
    __syncthreads();
    for (int s = 128; s > 0; s >>= 1) {
        if (tid < s) red[tid] = fmaxf(red[tid], red[tid + s]);
        __syncthreads();
    }
    mx = red[0];
    __syncthreads();

    float sum = 0.0f;
#pragma unroll
    for (int i = 0; i < 8; ++i) {
        v[i] = expf(v[i] - mx);
        sum += v[i];
    }
    red[tid] = sum;
    __syncthreads();
    for (int s = 128; s > 0; s >>= 1) {
        if (tid < s) red[tid] += red[tid + s];
        __syncthreads();
    }
    const float inv = 1.0f / red[0];
#pragma unroll
    for (int i = 0; i < 8; ++i) rp[tid + i * 256] = v[i] * inv;
}

extern "C" void kernel_launch(void* const* d_in, const int* in_sizes, int n_in,
                              void* d_out, int out_size, void* d_ws, size_t ws_size,
                              hipStream_t stream) {
    const float* F = (const float*)d_in[0];  // features   [32768, 512]
    const float* P = (const float*)d_in[1];  // prototypes [2048, 512]
    float* out = (float*)d_out;              // [32768, 2048]

    float* f2 = (float*)d_ws;                // 32768 floats
    float* p2 = f2 + NROWS;                  // 2048 floats
    const size_t norms_bytes = (size_t)(NROWS + KCLUS) * sizeof(float);
    const size_t planeF = (size_t)NROWS * DDIM;  // elements
    const size_t planeP = (size_t)KCLUS * DDIM;  // elements
    const size_t need = norms_bytes + 2 * (planeF + planeP) * sizeof(uint16_t);

    if (ws_size >= need) {
        uint16_t* Fhi = (uint16_t*)((char*)d_ws + norms_bytes);
        uint16_t* Flo = Fhi + planeF;
        uint16_t* Phi = Flo + planeF;
        uint16_t* Plo = Phi + planeP;

        // 1) one-shot split + norms
        convert_norms_kernel<<<(NROWS + KCLUS) / 8, 256, 0, stream>>>(
            F, P, f2, p2, Fhi, Flo, Phi, Plo);

        // 2) WMMA GEMM (pure loads + wmma) + fused similarity
        dim3 grid(KCLUS / 256, NROWS / 64);  // (8, 512)
        gemm_sim_planes_kernel<<<grid, 256, 0, stream>>>(
            Fhi, Flo, Phi, Plo, f2, p2, out);
    } else {
        norms_kernel<<<(NROWS + KCLUS) / 8, 256, 0, stream>>>(F, P, f2, p2);
        dim3 grid(KCLUS / 128, NROWS / 64);  // (16, 512)
        gemm_sim_inline_kernel<<<grid, 256, 0, stream>>>(F, P, f2, p2, out);
    }

    // 3) row softmax, one block per row
    softmax_kernel<<<NROWS, 256, 0, stream>>>(out);
}
